// PatchEncoder_78778290143693
// MI455X (gfx1250) — compile-verified, matrix operand order checked by reading, not run on hardware
//
#include <hip/hip_runtime.h>

typedef _Float16 v16h __attribute__((ext_vector_type(16)));
typedef float    v8f  __attribute__((ext_vector_type(8)));

#define TPB 256

// Total B-fragment elements: 7 ntiles * 24 rows * 6 ktiles * 32 lanes * 16 = 516096 f16 (~1 MB ws)
#define WFRAG_N (7 * 24 * 6 * 32 * 16)

// Pass 0: rewrite FC weights into f16 WMMA B-fragment order.
// Fragment layout: wfrag[ntile][r][kt][lane][pos], where lane's v16h holds
// K = kt*32 + (lane>=16 ? 8 : 0) + (pos&7) + (pos>=8 ? 16 : 0), N = ntile*16 + (lane&15),
// and the FC K axis is row-permuted: kk (0..191) -> c = kk/24, j = kk%24, k_orig = c*576 + r*24 + j.
__global__ void prep_fcw_frag(const float* __restrict__ fcw, _Float16* __restrict__ wfrag) {
  int i = blockIdx.x * TPB + threadIdx.x;
  if (i >= WFRAG_N) return;
  int pos = i & 15;
  int q1 = i >> 4;
  int lane = q1 & 31;
  int q2 = q1 >> 5;
  int kt = q2 % 6;
  int q3 = q2 / 6;
  int rr = q3 % 24;
  int ntile = q3 / 24;
  int n = ntile * 16 + (lane & 15);
  if (n > 99) n = 99;                       // tail cols: duplicate (never written out)
  int k32 = (pos & 7) + ((lane >> 4) << 3) + ((pos >> 3) << 4);
  int kk = kt * 32 + k32;                   // 0..191 within the row-chunk
  int c = kk / 24, j = kk - c * 24;
  wfrag[i] = (_Float16)fcw[n * 4608 + c * 576 + rr * 24 + j];
}

// Fused patch-extract + conv1/2/3 + WMMA FC, row-pipelined with ring buffers.
__launch_bounds__(TPB)
__global__ void patch_encoder(const float* __restrict__ latents,
                              const float* __restrict__ w1, const float* __restrict__ b1,
                              const float* __restrict__ w2, const float* __restrict__ b2,
                              const float* __restrict__ w3, const float* __restrict__ b3,
                              const _Float16* __restrict__ wfrag,
                              const float* __restrict__ fcb,
                              float* __restrict__ out) {
  __shared__ float w1s[36], b1s[4], w2s[288], b2s[8], w3s[576], b3s[8], fcbs[100];
  __shared__ long  pbase[16];
  __shared__ __align__(16) float    inpr[4][16][24];        //  6 KB input-row ring
  __shared__ __align__(16) _Float16 s1r[4][16][4][24];      // 12 KB conv1-row ring
  __shared__ __align__(16) _Float16 s2r[4][16][8][24];      // 24 KB conv2-row ring
  __shared__ __align__(16) _Float16 afrag[2][6][32][16];    // 12 KB A-fragments, dbl buffered

  const int tid = threadIdx.x;
  const int wg  = blockIdx.x;

  #pragma clang loop unroll(disable)
  for (int i = tid; i < 1020; i += TPB) {   // pack all small weight loads into one loop
    if (i < 36)        w1s[i]        = w1[i];
    else if (i < 40)   b1s[i - 36]   = b1[i - 36];
    else if (i < 328)  w2s[i - 40]   = w2[i - 40];
    else if (i < 336)  b2s[i - 328]  = b2[i - 328];
    else if (i < 912)  w3s[i - 336]  = w3[i - 336];
    else if (i < 920)  b3s[i - 912]  = b3[i - 912];
    else               fcbs[i - 920] = fcb[i - 920];
  }
  if (tid < 16) {
    // patch decode: loop order (segment, h, w, frame) per reference
    int gp = wg * 16 + tid;
    int v = gp / 1600, rem = gp - v * 1600;
    int s = rem / 100, rem2 = rem - s * 100;
    int hn = rem2 / 20, rem3 = rem2 - hn * 20;
    int wn = rem3 / 4, fp = rem3 - wn * 4;
    pbase[tid] = (((long)(v * 64 + s * 4 + fp)) * 120 + hn * 24) * 120 + wn * 24;
  }
  __syncthreads();

  const int lane = tid & 31, wv = tid >> 5;
  const int hi = lane >> 4, nrow = lane & 15;
  v8f acc = {};

  #pragma clang loop unroll(disable)
  for (int t = -4; t < 27; ++t) {
    // Stage L: load input row t+4
    int lr = t + 4;
    if (lr >= 0 && lr < 24) {
      #pragma clang loop unroll(disable)
      for (int idx = tid; idx < 16 * 24; idx += TPB) {
        int p = idx / 24, j = idx - p * 24;
        inpr[lr & 3][p][j] = latents[pbase[p] + (long)lr * 120 + j];
      }
    }
    // Stage 1: conv1 (1->4), output row t+2, reads inp rows t+1..t+3
    int r1 = t + 2;
    if (r1 >= 0 && r1 < 24) {
      #pragma clang loop unroll(disable)
      for (int idx = tid; idx < 16 * 4 * 24; idx += TPB) {
        int p = idx / 96, rm = idx - p * 96;
        int o = rm / 24, j = rm - o * 24;
        float a = b1s[o];
        #pragma unroll
        for (int di = 0; di < 3; ++di) {
          int ii = r1 + di - 1;
          if ((unsigned)ii < 24u) {
            #pragma unroll
            for (int dj = 0; dj < 3; ++dj) {
              int jj = j + dj - 1;
              if ((unsigned)jj < 24u)
                a += w1s[o * 9 + di * 3 + dj] * inpr[ii & 3][p][jj];
            }
          }
        }
        s1r[r1 & 3][p][o][j] = (_Float16)fmaxf(a, 0.f);
      }
    }
    // Stage 2: conv2 (4->8), output row t, reads s1 rows t-1..t+1
    if (t >= 0 && t < 24) {
      #pragma clang loop unroll(disable)
      for (int idx = tid; idx < 16 * 8 * 24; idx += TPB) {
        int p = idx / 192, rm = idx - p * 192;
        int o = rm / 24, j = rm - o * 24;
        float a = b2s[o];
        #pragma clang loop unroll(disable)
        for (int ci = 0; ci < 4; ++ci) {
          #pragma unroll
          for (int di = 0; di < 3; ++di) {
            int ii = t + di - 1;
            if ((unsigned)ii < 24u) {
              #pragma unroll
              for (int dj = 0; dj < 3; ++dj) {
                int jj = j + dj - 1;
                if ((unsigned)jj < 24u)
                  a += w2s[(o * 4 + ci) * 9 + di * 3 + dj] * (float)s1r[ii & 3][p][ci][jj];
              }
            }
          }
        }
        s2r[t & 3][p][o][j] = (_Float16)fmaxf(a, 0.f);
      }
    }
    // Stage 3: conv3 (8->8), output row t-2, scattered into A-fragment layout
    int r3 = t - 2;
    if (r3 >= 0 && r3 < 24) {
      #pragma clang loop unroll(disable)
      for (int idx = tid; idx < 16 * 8 * 24; idx += TPB) {
        int p = idx / 192, rm = idx - p * 192;
        int o = rm / 24, j = rm - o * 24;
        float a = b3s[o];
        #pragma clang loop unroll(disable)
        for (int ci = 0; ci < 8; ++ci) {
          #pragma unroll
          for (int di = 0; di < 3; ++di) {
            int ii = r3 + di - 1;
            if ((unsigned)ii < 24u) {
              #pragma unroll
              for (int dj = 0; dj < 3; ++dj) {
                int jj = j + dj - 1;
                if ((unsigned)jj < 24u)
                  a += w3s[(o * 8 + ci) * 9 + di * 3 + dj] * (float)s2r[ii & 3][p][ci][jj];
              }
            }
          }
        }
        int kk = o * 24 + j;                        // K within this row-chunk, 0..191
        int kt = kk >> 5, k32 = kk & 31;
        int dlane = p + ((k32 >> 3) & 1) * 16;      // hi/lo half-wave
        int dpos = (k32 & 7) + ((k32 >> 4) << 3);   // element within lane's v16h
        afrag[r3 & 1][kt][dlane][dpos] = (_Float16)fmaxf(a, 0.f);
      }
    }
    // Stage 4: WMMA FC on fragment row t-3 (6 chained K-tiles of 32)
    int r4 = t - 3;
    if (r4 >= 0 && wv < 7) {                        // wave-uniform: EXEC all-ones inside
      const _Float16* ar = &afrag[r4 & 1][0][lane][0];
      const _Float16* wr = wfrag + (long)(wv * 24 + r4) * 3072 + lane * 16;
      #pragma unroll
      for (int kt = 0; kt < 6; ++kt) {
        v16h A = *(const v16h*)(ar + kt * 512);
        v16h B = *(const v16h*)(wr + kt * 512);
        acc = __builtin_amdgcn_wmma_f32_16x16x32_f16(false, A, false, B,
                                                     (short)0, acc, false, false);
      }
    }
    __syncthreads();                                // retire one pipeline step
  }

  // epilogue: C/D layout -> VGPR r holds M=r (lanes<16) / M=r+8 (lanes>=16), N=lane&15
  if (wv < 7) {
    int neuron = wv * 16 + nrow;
    if (neuron < 100) {
      float bias = fcbs[neuron];
      #pragma unroll
      for (int r = 0; r < 8; ++r) {
        long gp = (long)wg * 16 + r + hi * 8;
        out[gp * 100 + neuron] = acc[r] + bias;
      }
    }
  }
}

extern "C" void kernel_launch(void* const* d_in, const int* in_sizes, int n_in,
                              void* d_out, int out_size, void* d_ws, size_t ws_size,
                              hipStream_t stream) {
  const float* latents = (const float*)d_in[0];
  const float* w1  = (const float*)d_in[1];
  const float* b1  = (const float*)d_in[2];
  const float* w2  = (const float*)d_in[3];
  const float* b2  = (const float*)d_in[4];
  const float* w3  = (const float*)d_in[5];
  const float* b3  = (const float*)d_in[6];
  const float* fcw = (const float*)d_in[7];
  const float* fcb = (const float*)d_in[8];
  float* out = (float*)d_out;
  _Float16* wfrag = (_Float16*)d_ws;    // ~1.0 MB scratch for B-fragments

  prep_fcw_frag<<<dim3((WFRAG_N + TPB - 1) / TPB), dim3(TPB), 0, stream>>>(fcw, wfrag);
  patch_encoder<<<dim3(800), dim3(TPB), 0, stream>>>(latents, w1, b1, w2, b2, w3, b3,
                                                     (const _Float16*)wfrag, fcb, out);
}
